// VQCodebook_26603027431775
// MI455X (gfx1250) — matrix operand order repaired and checked by reading, not compiled, not run on hardware
//
#include <hip/hip_runtime.h>
#include <hip/hip_bf16.h>
#include <stdint.h>

#define N_VOCAB   8192
#define D_EMB     256
#define TOKENS    16384              // 16 * 1024
#define NPART     4
#define PART_V    (N_VOCAB / NPART)  // 2048 codes per vocab partition
#define CHUNK     32                 // codebook rows per LDS buffer
#define NCHUNK    (PART_V / CHUNK)   // 64 double-buffered iterations
#define LDS_PAD   8                  // ushort pad per row -> 528B row stride
#define LDS_STRIDE (D_EMB + LDS_PAD)
#define NWAVES    8
#define WGT       256
#define COMMIT    0.25f

typedef __attribute__((ext_vector_type(16))) __bf16 v16bf;
typedef __attribute__((ext_vector_type(8)))  __bf16 v8bf;
typedef __attribute__((ext_vector_type(8)))  float  v8f;

// ---------------------------------------------------------------------------
// Kernel 1: codebook f32 -> bf16 (workspace) + per-row squared norms.
// ---------------------------------------------------------------------------
__global__ void __launch_bounds__(WGT)
vq_prep_cb(const float* __restrict__ cb, uint16_t* __restrict__ cbh,
           float* __restrict__ cbnorm) {
    int wave = (blockIdx.x * WGT + threadIdx.x) >> 5;
    int lane = threadIdx.x & 31;
    if (wave >= N_VOCAB) return;
    const float* src = cb + (size_t)wave * D_EMB;
    uint16_t*    dst = cbh + (size_t)wave * D_EMB;
    float s = 0.f;
#pragma unroll
    for (int k = lane; k < D_EMB; k += 32) {
        float f = src[k];
        s += f * f;
        __bf16 h = (__bf16)f;
        dst[k] = __builtin_bit_cast(uint16_t, h);
    }
#pragma unroll
    for (int m = 16; m; m >>= 1) s += __shfl_xor(s, m, 32);
    if (lane == 0) cbnorm[wave] = s;
}

// ---------------------------------------------------------------------------
// Kernel 2: WMMA bf16 distance GEMM + running argmin.
// Each wave: 16 tokens (A resident in 64 VGPRs). Vocab partition streamed
// through a double-buffered LDS tile (async global->LDS b128 copies overlap
// the WMMA chain of the other buffer).
// ---------------------------------------------------------------------------
__global__ void __launch_bounds__(WGT)
vq_argmin(const float* __restrict__ ze, const uint16_t* __restrict__ cbh,
          const float* __restrict__ cbnorm,
          float* __restrict__ part_val, int* __restrict__ part_idx) {
    __shared__ uint16_t lcb[2][CHUNK * LDS_STRIDE];   // 2 x 32 x 528B = 33 KB

    const int tid  = threadIdx.x;
    const int wave = tid >> 5;
    const int lane = tid & 31;
    const int lrow = lane & 15;        // row/col within 16-lane half
    const int lhi  = lane >> 4;        // 0: lanes 0-15, 1: lanes 16-31

    const int tokBase = blockIdx.x * (NWAVES * 16) + wave * 16;
    const int part    = blockIdx.y;
    const int vBase   = part * PART_V;

    // ---- A fragments: token rows, f32 -> bf16, ISA 16-bit A 16x32 layout:
    // lane holds M = lane%16; K = c0+{0..7} and c0+16..c0+23 per K-tile,
    // with c0 = 8 for the upper half-wave.
    v16bf afrag[8];
    {
        const float* arow = ze + (size_t)(tokBase + lrow) * D_EMB;
        const int c0 = lhi * 8;
#pragma unroll
        for (int kt = 0; kt < 8; ++kt) {
            const float* p = arow + kt * 32 + c0;
            float4 f0 = *(const float4*)(p);
            float4 f1 = *(const float4*)(p + 4);
            float4 f2 = *(const float4*)(p + 16);
            float4 f3 = *(const float4*)(p + 20);
            v16bf a;
            a[0]  = (__bf16)f0.x; a[1]  = (__bf16)f0.y; a[2]  = (__bf16)f0.z; a[3]  = (__bf16)f0.w;
            a[4]  = (__bf16)f1.x; a[5]  = (__bf16)f1.y; a[6]  = (__bf16)f1.z; a[7]  = (__bf16)f1.w;
            a[8]  = (__bf16)f2.x; a[9]  = (__bf16)f2.y; a[10] = (__bf16)f2.z; a[11] = (__bf16)f2.w;
            a[12] = (__bf16)f3.x; a[13] = (__bf16)f3.y; a[14] = (__bf16)f3.z; a[15] = (__bf16)f3.w;
            afrag[kt] = a;
        }
    }

    float bestv[8];
    int   besti[8];
#pragma unroll
    for (int r = 0; r < 8; ++r) { bestv[r] = 3.4e38f; besti[r] = 0; }

    const int stageRow = tid >> 6;            // 4 rows per pass (256 thr / 64)
    const int stageCol = tid & 63;            // 2 x 16B cols... (see below)
    (void)stageRow; (void)stageCol;

    // Stage one CHUNK (32 rows x 512B = 16KB): 1024 x 16B segs / 256 thr = 4.
    auto stage = [&](int buf, int ch) {
        const uint16_t* src = cbh + (size_t)(vBase + ch) * D_EMB;
        const uint32_t  ldsBase = (uint32_t)(uintptr_t)(&lcb[buf][0]);
#pragma unroll
        for (int pass = 0; pass < 4; ++pass) {
            int e   = pass * WGT + tid;
            int row = e >> 5;
            int col = e & 31;                                // 16B column
            uint32_t dst  = ldsBase + (uint32_t)(row * (LDS_STRIDE * 2) + col * 16);
            uint32_t soff = (uint32_t)(row * (D_EMB * 2) + col * 16);
            asm volatile("global_load_async_to_lds_b128 %0, %1, %2"
                         :: "v"(dst), "v"(soff), "s"(src) : "memory");
        }
    };

    stage(0, 0);                                             // prologue

    for (int it = 0; it < NCHUNK; ++it) {
        const int cur = it & 1;
        const int ch  = it * CHUNK;

        asm volatile("s_wait_asynccnt 0" ::: "memory");      // cur buffer landed
        __syncthreads();                                     // all waves: landed,
                                                             // and other buf free
        if (it + 1 < NCHUNK) stage(cur ^ 1, ch + CHUNK);     // overlap next DMA

        // hoist cbnorm loads so they overlap the WMMA chains
        const int n0 = vBase + ch + lrow;
        const int n1 = n0 + 16;
        const float cn0 = cbnorm[n0];
        const float cn1 = cbnorm[n1];

#pragma unroll
        for (int nt = 0; nt < CHUNK / 16; ++nt) {
            // B 32x16 bf16 layout: lane holds N = lane%16; K = c0+{0..7} and
            // c0+16..c0+23 per K-tile, with c0 = 8 for lanes 16-31 (matches A).
            const uint16_t* bbase =
                &lcb[cur][(nt * 16 + lrow) * LDS_STRIDE + lhi * 8];

            v16bf bf[8];
#pragma unroll
            for (int kt = 0; kt < 8; ++kt) {                 // 16 ds_load_b128
                v8bf lo = *(const v8bf*)(bbase + kt * 32);
                v8bf hi = *(const v8bf*)(bbase + kt * 32 + 16);
                bf[kt] = __builtin_shufflevector(lo, hi,
                          0,1,2,3,4,5,6,7,8,9,10,11,12,13,14,15);
            }
            v8f acc = {0.f, 0.f, 0.f, 0.f, 0.f, 0.f, 0.f, 0.f};
#pragma unroll
            for (int kt = 0; kt < 8; ++kt)                   // 8 chained WMMAs
                acc = __builtin_amdgcn_wmma_f32_16x16x32_bf16(
                        false, afrag[kt], false, bf[kt], (short)0, acc,
                        false, false);

            const int   n  = (nt == 0) ? n0 : n1;
            const float cn = (nt == 0) ? cn0 : cn1;
#pragma unroll
            for (int r = 0; r < 8; ++r) {
                float d = cn - 2.0f * acc[r];   // + ||z||^2 const per token
                if (d < bestv[r]) { bestv[r] = d; besti[r] = n; }
            }
        }
    }

    // Reduce across the 16 lanes of each half-wave (codes n%16) per token.
#pragma unroll
    for (int r = 0; r < 8; ++r) {
        float v = bestv[r];
        int   i = besti[r];
#pragma unroll
        for (int m = 8; m; m >>= 1) {
            float ov = __shfl_xor(v, m, 32);
            int   oi = __shfl_xor(i, m, 32);
            if (ov < v || (ov == v && oi < i)) { v = ov; i = oi; }
        }
        bestv[r] = v; besti[r] = i;
    }
    if (lrow == 0) {
        int tok = tokBase + lhi * 8;   // lane0 -> tokens 0..7, lane16 -> 8..15
#pragma unroll
        for (int r = 0; r < 8; ++r) {
            part_val[(size_t)(tok + r) * NPART + part] = bestv[r];
            part_idx[(size_t)(tok + r) * NPART + part] = besti[r];
        }
    }
}

// ---------------------------------------------------------------------------
// Kernel 3: combine vocab partitions, gather z_q (exact f32), emit indices,
// and deterministic per-block loss partial sums.
// ---------------------------------------------------------------------------
__global__ void __launch_bounds__(WGT)
vq_gather(const float* __restrict__ ze, const float* __restrict__ cb,
          const float* __restrict__ part_val, const int* __restrict__ part_idx,
          float* __restrict__ out_zq, float* __restrict__ out_idx,
          float* __restrict__ loss_part) {
    __shared__ float wsum[NWAVES];
    const int wave = threadIdx.x >> 5;
    const int lane = threadIdx.x & 31;
    const int tok  = blockIdx.x * NWAVES + wave;

    int idx = 0;
    if (lane == 0) {
        float bv = part_val[(size_t)tok * NPART];
        idx      = part_idx[(size_t)tok * NPART];
#pragma unroll
        for (int p = 1; p < NPART; ++p) {
            float v = part_val[(size_t)tok * NPART + p];
            int   i = part_idx[(size_t)tok * NPART + p];
            if (v < bv || (v == bv && i < idx)) { bv = v; idx = i; }
        }
        out_idx[tok] = (float)idx;
    }
    idx = __shfl(idx, 0, 32);

    const float* crow = cb + (size_t)idx * D_EMB;
    const float* zrow = ze + (size_t)tok * D_EMB;
    float*       orow = out_zq + (size_t)tok * D_EMB;
    float s = 0.f;
#pragma unroll
    for (int k = lane; k < D_EMB; k += 32) {
        float q = crow[k];
        float d = q - zrow[k];
        s += d * d;
        orow[k] = q;                 // z_q_st value == z_q
    }
#pragma unroll
    for (int m = 16; m; m >>= 1) s += __shfl_xor(s, m, 32);
    if (lane == 0) wsum[wave] = s;
    __syncthreads();
    if (threadIdx.x == 0) {
        float t = 0.f;
#pragma unroll
        for (int w = 0; w < NWAVES; ++w) t += wsum[w];
        loss_part[blockIdx.x] = t;
    }
}

// ---------------------------------------------------------------------------
// Kernel 4: fixed-order final loss reduction (deterministic, no atomics).
// ---------------------------------------------------------------------------
__global__ void __launch_bounds__(WGT)
vq_loss_final(const float* __restrict__ loss_part, float* __restrict__ out_loss) {
    __shared__ float part[WGT];
    float s = 0.f;
#pragma unroll
    for (int i = 0; i < 8; ++i) s += loss_part[threadIdx.x * 8 + i];
    part[threadIdx.x] = s;
    __syncthreads();
    if (threadIdx.x == 0) {
        float t = 0.f;
        for (int i = 0; i < WGT; ++i) t += part[i];
        out_loss[0] = (1.0f + COMMIT) * t / (float)((size_t)TOKENS * D_EMB);
    }
}

// ---------------------------------------------------------------------------
extern "C" void kernel_launch(void* const* d_in, const int* in_sizes, int n_in,
                              void* d_out, int out_size, void* d_ws, size_t ws_size,
                              hipStream_t stream) {
    const float* ze = (const float*)d_in[0];   // (16,1024,256) f32
    const float* cb = (const float*)d_in[1];   // (8192,256)    f32
    float* out = (float*)d_out;

    // workspace layout (bytes)
    char* ws = (char*)d_ws;
    uint16_t* cbh       = (uint16_t*)(ws + 0);          // 4,194,304 B
    float*    cbnorm    = (float*)   (ws + 4194304);    //    32,768 B
    float*    part_val  = (float*)   (ws + 4227072);    //   262,144 B
    int*      part_idx  = (int*)     (ws + 4489216);    //   262,144 B
    float*    loss_part = (float*)   (ws + 4751360);    //     8,192 B

    float* out_zq   = out;                              // 4,194,304 floats
    float* out_idx  = out + (size_t)TOKENS * D_EMB;     //    16,384 floats
    float* out_loss = out_idx + TOKENS;                 //         1 float

    vq_prep_cb<<<N_VOCAB / NWAVES, WGT, 0, stream>>>(cb, cbh, cbnorm);

    dim3 grid(TOKENS / (NWAVES * 16), NPART);           // (128, 4)
    vq_argmin<<<grid, WGT, 0, stream>>>(ze, cbh, cbnorm, part_val, part_idx);

    vq_gather<<<TOKENS / NWAVES, WGT, 0, stream>>>(ze, cb, part_val, part_idx,
                                                   out_zq, out_idx, loss_part);

    vq_loss_final<<<1, WGT, 0, stream>>>(loss_part, out_loss);
}
